// GeneralizedCrossMessage_81363860456160
// MI455X (gfx1250) — matrix operand order repaired and checked by reading, not compiled, not run on hardware
//
#include <hip/hip_runtime.h>
#include <hip/hip_bf16.h>
#include <math.h>

typedef _Float16 v16h __attribute__((ext_vector_type(16)));
typedef _Float16 v8h  __attribute__((ext_vector_type(8)));
typedef float    v8f  __attribute__((ext_vector_type(8)));
typedef float    v4f  __attribute__((ext_vector_type(4)));

#define SDIM 1024
#define HDIM 768
#define DD   24
#define MHD  96
#define FEAT 96      // 4*D
#define FIN  2304    // 3*H
#define TPASS 512    // t rows per attention pass
#define FSTR  104    // feat row stride in halves (208B: 16B-aligned, bank-conflict-free)
#define NEG_BIG (-3.402823466e38f)

// ---------------------------------------------------------------------------
// Zj = H_j @ Wj, Zi = H_i @ Wi   (tiny: 0.075 GF total)
// ---------------------------------------------------------------------------
__global__ __launch_bounds__(64)
void proj_kernel(const float* __restrict__ H_j, const float* __restrict__ H_i,
                 const float* __restrict__ Wj, const float* __restrict__ Wi,
                 float* __restrict__ Zj, float* __restrict__ Zi)
{
    __shared__ __align__(16) float rowJ[HDIM];
    __shared__ __align__(16) float rowI[HDIM];
    const int s = blockIdx.x, tid = threadIdx.x;
    for (int i = tid * 4; i < HDIM; i += 64 * 4) {
        *(v4f*)&rowJ[i] = *(const v4f*)&H_j[s * HDIM + i];
        *(v4f*)&rowI[i] = *(const v4f*)&H_i[s * HDIM + i];
    }
    __syncthreads();
    if (tid < DD) {
        float a = 0.f;
        for (int h = 0; h < HDIM; ++h) a += rowJ[h] * Wj[h * DD + tid];
        Zj[s * DD + tid] = a;
    } else if (tid >= 32 && tid < 32 + DD) {
        const int d = tid - 32;
        float a = 0.f;
        for (int h = 0; h < HDIM; ++h) a += rowI[h] * Wi[h * DD + d];
        Zi[s * DD + d] = a;
    }
}

// ---------------------------------------------------------------------------
// f32 -> f16 conversion
// ---------------------------------------------------------------------------
__global__ void cvt_f32_f16(const float* __restrict__ x, _Float16* __restrict__ y, int n)
{
    const int i = blockIdx.x * blockDim.x + threadIdx.x;
    if (i < n) y[i] = (_Float16)x[i];
}

// ---------------------------------------------------------------------------
// msg_in = [ctx, H_j, ctx*H_j]  (f16)
// ---------------------------------------------------------------------------
__global__ void build_msgin(const float* __restrict__ ctx, const float* __restrict__ H_j,
                            _Float16* __restrict__ msgin)
{
    const int i = blockIdx.x * blockDim.x + threadIdx.x;
    if (i >= SDIM * FIN) return;
    const int s = i / FIN, f = i % FIN;
    float v;
    if (f < HDIM)            v = ctx[s * HDIM + f];
    else if (f < 2 * HDIM)   v = H_j[s * HDIM + (f - HDIM)];
    else {
        const int h = f - 2 * HDIM;
        v = ctx[s * HDIM + h] * H_j[s * HDIM + h];
    }
    msgin[i] = (_Float16)v;
}

// ---------------------------------------------------------------------------
// Fused attention: per block = one query row s.
//   logits[t] = W2 . relu( pair(s,t) @ W1 + b1 ) + b2   via v_wmma_f32_16x16x32_f16
//   probs[s,:] = softmax(logits)  (f16 out)
// Two passes of 512 t-rows: pass builds an f16 feature table in LDS (each
// element computed exactly once, v8h-vectorized), then waves stream A
// fragments out of it with ds_load_b128 pairs. W1 is packed once into LDS in
// B-fragment-linear order (2x16B ds loads per fragment).
// ---------------------------------------------------------------------------
__global__ __launch_bounds__(256)
void attn_kernel(const float* __restrict__ Zj, const float* __restrict__ Zi,
                 const float* __restrict__ attn_mask,
                 const float* __restrict__ W1, const float* __restrict__ b1,
                 const float* __restrict__ W2, const float* __restrict__ b2,
                 _Float16* __restrict__ probs)
{
    __shared__ __align__(16) _Float16 feat[TPASS][FSTR];  // ~104 KB
    __shared__ __align__(16) _Float16 W1p[18][512];       // 18 KB packed B frags
    __shared__ __align__(16) _Float16 zjh[DD + 8];        // f16 Zj row (padded)
    __shared__ float b1s[MHD];
    __shared__ float w2s[MHD];
    __shared__ float zjs_f[DD];
    __shared__ float lg[SDIM];
    __shared__ float red[16];

    const int s    = blockIdx.x;
    const int tid  = threadIdx.x;
    const int lane = tid & 31;
    const int w    = tid >> 5;

    if (tid < DD) {
        const float zv = Zj[s * DD + tid];
        zjs_f[tid] = zv;
        zjh[tid]   = (_Float16)zv;
    }
    if (tid >= DD && tid < DD + 8) zjh[tid] = (_Float16)0.f;
    if (tid < MHD) { b1s[tid] = b1[tid]; w2s[tid] = W2[tid]; }
    for (int idx = tid; idx < FEAT * MHD; idx += 256) {
        const int k = idx / MHD, n = idx % MHD;
        const int c = k >> 5, k32 = k & 31;
        const int l = (n & 15) + ((k32 & 8) ? 16 : 0);
        const int e = (k32 & 7) + ((k32 & 16) ? 8 : 0);
        W1p[c * 6 + (n >> 4)][l * 16 + e] = (_Float16)W1[idx];
    }
    __syncthreads();

    const float b2v = b2[0];

    for (int pass = 0; pass < 2; ++pass) {
        const int tb = pass * TPASS;

        // ---- build feature table: feat[r][k], k in [0,96) ----
        for (int r = tid; r < TPASS; r += 256) {
            const float* zi = &Zi[(tb + r) * DD];
            float z[DD];
            #pragma unroll
            for (int j = 0; j < DD; j += 4) *(v4f*)&z[j] = *(const v4f*)&zi[j];
            _Float16* out = &feat[r][0];
            *(v8h*)&out[0]  = *(const v8h*)&zjh[0];
            *(v8h*)&out[8]  = *(const v8h*)&zjh[8];
            *(v8h*)&out[16] = *(const v8h*)&zjh[16];
            #pragma unroll
            for (int j0 = 0; j0 < DD; j0 += 8) {
                union { _Float16 e[8]; v8h v; } a, p, d;
                #pragma unroll
                for (int j = 0; j < 8; ++j) {
                    const float zjv = zjs_f[j0 + j];
                    const float ziv = z[j0 + j];
                    a.e[j] = (_Float16)ziv;
                    p.e[j] = (_Float16)(zjv * ziv);
                    // |zj - zi| as max(a-b, b-a): avoids packed-f32 fabs
                    // pattern that crashes gfx1250 ISel in this toolchain.
                    d.e[j] = (_Float16)fmaxf(zjv - ziv, ziv - zjv);
                }
                *(v8h*)&out[24 + j0] = a.v;
                *(v8h*)&out[48 + j0] = p.v;
                *(v8h*)&out[72 + j0] = d.v;
            }
        }
        __syncthreads();

        // ---- WMMA over this pass's 32 tiles ----
        for (int tile = w; tile < TPASS / 16; tile += 8) {
            const int t0r = tile * 16;           // row within feat
            const int t0  = tb + t0r;            // global t
            const int m    = lane & 15;
            const int koff = (lane < 16) ? 0 : 8;
            const _Float16* ar = &feat[t0r + m][0];
            v16h fa[3];
            #pragma unroll
            for (int c = 0; c < 3; ++c) {
                union { v16h v; v8h h[2]; } ua;
                ua.h[0] = *(const v8h*)&ar[c * 32 + koff];
                ua.h[1] = *(const v8h*)&ar[c * 32 + 16 + koff];
                fa[c] = ua.v;
            }
            float pl[8];
            #pragma unroll
            for (int v = 0; v < 8; ++v) pl[v] = 0.f;
            const int col = lane & 15;
            #pragma unroll
            for (int nt = 0; nt < 6; ++nt) {
                v8f acc = {};
                #pragma unroll
                for (int c = 0; c < 3; ++c) {
                    union { v16h v; v8h h[2]; } ub;
                    const _Float16* bp = &W1p[c * 6 + nt][lane * 16];
                    ub.h[0] = *(const v8h*)bp;
                    ub.h[1] = *(const v8h*)(bp + 8);
                    acc = __builtin_amdgcn_wmma_f32_16x16x32_f16(
                              false, fa[c], false, ub.v, (short)0, acc, false, false);
                }
                const float bb = b1s[nt * 16 + col];
                const float ww = w2s[nt * 16 + col];
                #pragma unroll
                for (int v = 0; v < 8; ++v) {
                    float h = acc[v] + bb;
                    h = fmaxf(h, 0.f);
                    pl[v] += h * ww;
                }
            }
            #pragma unroll
            for (int v = 0; v < 8; ++v) {
                float x = pl[v];
                x += __shfl_xor(x, 1, 32);
                x += __shfl_xor(x, 2, 32);
                x += __shfl_xor(x, 4, 32);
                x += __shfl_xor(x, 8, 32);
                pl[v] = x;
            }
            if ((lane & 15) == 0) {
                const int mm = (lane >> 4) * 8;
                #pragma unroll
                for (int v = 0; v < 8; ++v) {
                    const int t = t0 + mm + v;
                    lg[t] = pl[v] + b2v + (1.0f - attn_mask[t]) * NEG_BIG;
                }
            }
        }
        __syncthreads();   // protect feat before next pass rebuild
    }

    // ---- softmax over 1024 logits ----
    float mx = NEG_BIG;
    for (int i = tid; i < SDIM; i += 256) mx = fmaxf(mx, lg[i]);
    for (int off = 16; off >= 1; off >>= 1) mx = fmaxf(mx, __shfl_xor(mx, off, 32));
    if (lane == 0) red[w] = mx;
    __syncthreads();
    if (tid == 0) {
        float m2 = red[0];
        for (int i = 1; i < 8; ++i) m2 = fmaxf(m2, red[i]);
        red[8] = m2;
    }
    __syncthreads();
    mx = red[8];
    float sum = 0.f;
    for (int i = tid; i < SDIM; i += 256) sum += __expf(lg[i] - mx);
    for (int off = 16; off >= 1; off >>= 1) sum += __shfl_xor(sum, off, 32);
    if (lane == 0) red[w] = sum;
    __syncthreads();
    if (tid == 0) {
        float s2 = 0.f;
        for (int i = 0; i < 8; ++i) s2 += red[i];
        red[9] = s2;
    }
    __syncthreads();
    const float inv = 1.0f / red[9];
    for (int i = tid; i < SDIM; i += 256)
        probs[(size_t)s * SDIM + i] = (_Float16)(__expf(lg[i] - mx) * inv);
}

// ---------------------------------------------------------------------------
// Generic f16 WMMA GEMM: C[M,N] = epi(A[M,K] @ B[K,N])
// BM=64, BN=128, BK=32; 256 threads = 8 waves (4 M-subtiles x 2 N-halves).
// B staged into LDS in fragment-linear order; A staged row-major (stride 40).
// Next K-chunk prefetched with global_prefetch.
// ---------------------------------------------------------------------------
__global__ __launch_bounds__(256)
void gemm_f16_kernel(const _Float16* __restrict__ A, const _Float16* __restrict__ B,
                     int M, int N, int K,
                     const float* __restrict__ bias, const float* __restrict__ alpha_p,
                     int relu, float* __restrict__ Cf, _Float16* __restrict__ Ch)
{
    __shared__ __align__(16) _Float16 As[64 * 40];  // 64x32, stride 40 halves
    __shared__ __align__(16) _Float16 Bp[8][512];   // packed frags per 16-col N tile

    const int tid  = threadIdx.x;
    const int lane = tid & 31;
    const int w    = tid >> 5;
    const int wm   = w & 3;
    const int wn   = w >> 2;
    const int m0   = blockIdx.y * 64;
    const int n0   = blockIdx.x * 128;

    v8f acc[4] = {};

    for (int k0 = 0; k0 < K; k0 += 32) {
        __syncthreads();
        {   // stage A: 8 contiguous halves (16B) per thread
            const int r = tid >> 2, seg = tid & 3;
            *(v8h*)&As[r * 40 + seg * 8] =
                *(const v8h*)&A[(size_t)(m0 + r) * K + k0 + seg * 8];
        }
        {   // stage B: 16 halves per thread, scattered into fragment order
            const int r     = tid >> 3;       // k row in chunk, 0..31
            const int cseg  = tid & 7;        // which 16-col N tile
            const int lhalf = (r & 8) ? 16 : 0;
            const int e     = (r & 7) + ((r & 16) ? 8 : 0);
            const _Float16* gb = &B[(size_t)(k0 + r) * N + n0 + cseg * 16];
            #pragma unroll
            for (int c = 0; c < 16; ++c)
                Bp[cseg][(c + lhalf) * 16 + e] = gb[c];
        }
        if (k0 + 32 < K) {  // prefetch next chunk (global_prefetch_b8)
            __builtin_prefetch(&A[(size_t)(m0 + (tid >> 2)) * K + k0 + 32], 0, 0);
            __builtin_prefetch(&B[(size_t)(k0 + 32 + (tid >> 3)) * N + n0 + (tid & 7) * 16], 0, 0);
        }
        __syncthreads();

        union { v16h v; v8h h[2]; } ua;
        {
            const _Float16* ap =
                &As[(wm * 16 + (lane & 15)) * 40 + ((lane < 16) ? 0 : 8)];
            ua.h[0] = *(const v8h*)ap;
            ua.h[1] = *(const v8h*)(ap + 16);
        }
        #pragma unroll
        for (int nt = 0; nt < 4; ++nt) {
            union { v16h v; v8h h[2]; } ub;
            const _Float16* bp = &Bp[wn * 4 + nt][lane * 16];
            ub.h[0] = *(const v8h*)bp;
            ub.h[1] = *(const v8h*)(bp + 8);
            acc[nt] = __builtin_amdgcn_wmma_f32_16x16x32_f16(
                          false, ua.v, false, ub.v, (short)0, acc[nt], false, false);
        }
    }

    const float alpha = alpha_p ? alpha_p[0] : 1.0f;
    #pragma unroll
    for (int nt = 0; nt < 4; ++nt) {
        const int gn = n0 + wn * 64 + nt * 16 + (lane & 15);
        const float bv = bias ? bias[gn] : 0.0f;
        #pragma unroll
        for (int v = 0; v < 8; ++v) {
            const int gm = m0 + wm * 16 + (lane >> 4) * 8 + v;
            float r = acc[nt][v] + bv;
            if (relu) r = fmaxf(r, 0.f);
            r *= alpha;
            if (Cf) Cf[(size_t)gm * N + gn] = r;
            if (Ch) Ch[(size_t)gm * N + gn] = (_Float16)r;
        }
    }
}

// ---------------------------------------------------------------------------
// Host orchestration
// ---------------------------------------------------------------------------
extern "C" void kernel_launch(void* const* d_in, const int* in_sizes, int n_in,
                              void* d_out, int out_size, void* d_ws, size_t ws_size,
                              hipStream_t stream)
{
    const float* H_j  = (const float*)d_in[0];
    const float* H_i  = (const float*)d_in[1];
    const float* mask = (const float*)d_in[2];
    const float* Wj   = (const float*)d_in[3];
    const float* Wi   = (const float*)d_in[4];
    const float* W1   = (const float*)d_in[5];
    const float* b1   = (const float*)d_in[6];
    const float* W2   = (const float*)d_in[7];
    const float* b2   = (const float*)d_in[8];
    const float* V1   = (const float*)d_in[9];
    const float* bv1  = (const float*)d_in[10];
    const float* V2   = (const float*)d_in[11];
    const float* bv2  = (const float*)d_in[12];
    const float* alpha= (const float*)d_in[13];

    char* ws = (char*)d_ws;
    size_t off = 0;
    auto alloc = [&](size_t bytes) { char* p = ws + off; off += (bytes + 255) & ~(size_t)255; return p; };

    float*    Zj      = (float*)   alloc((size_t)SDIM * DD * 4);
    float*    Zi      = (float*)   alloc((size_t)SDIM * DD * 4);
    float*    ctx     = (float*)   alloc((size_t)SDIM * HDIM * 4);
    _Float16* Hi16    = (_Float16*)alloc((size_t)SDIM * HDIM * 2);
    _Float16* probs16 = (_Float16*)alloc((size_t)SDIM * SDIM * 2);
    _Float16* msgin16 = (_Float16*)alloc((size_t)SDIM * FIN * 2);
    _Float16* hv16    = (_Float16*)alloc((size_t)SDIM * HDIM * 2);
    _Float16* V1h     = (_Float16*)alloc((size_t)FIN * HDIM * 2);
    _Float16* V2h     = (_Float16*)alloc((size_t)HDIM * HDIM * 2);
    (void)ws_size; (void)n_in; (void)in_sizes; (void)out_size;

    // 1. projections
    proj_kernel<<<SDIM, 64, 0, stream>>>(H_j, H_i, Wj, Wi, Zj, Zi);

    // 2. f16 operand copies
    {
        int n = SDIM * HDIM;
        cvt_f32_f16<<<(n + 255) / 256, 256, 0, stream>>>(H_i, Hi16, n);
        n = FIN * HDIM;
        cvt_f32_f16<<<(n + 255) / 256, 256, 0, stream>>>(V1, V1h, n);
        n = HDIM * HDIM;
        cvt_f32_f16<<<(n + 255) / 256, 256, 0, stream>>>(V2, V2h, n);
    }

    // 3. fused pairwise-MLP attention scores + softmax (dominant 19.3 GF on WMMA)
    attn_kernel<<<SDIM, 256, 0, stream>>>(Zj, Zi, mask, W1, b1, W2, b2, probs16);

    // 4. ctx = probs @ H_i   (M=1024, N=768, K=1024)
    gemm_f16_kernel<<<dim3(HDIM / 128, SDIM / 64), 256, 0, stream>>>(
        probs16, Hi16, SDIM, HDIM, SDIM, nullptr, nullptr, 0, ctx, nullptr);

    // 5. msg_in = [ctx, H_j, ctx*H_j] in f16
    {
        int n = SDIM * FIN;
        build_msgin<<<(n + 255) / 256, 256, 0, stream>>>(ctx, H_j, msgin16);
    }

    // 6. hv = relu(msg_in @ V1 + bv1)   (M=1024, N=768, K=2304)
    gemm_f16_kernel<<<dim3(HDIM / 128, SDIM / 64), 256, 0, stream>>>(
        msgin16, V1h, SDIM, HDIM, FIN, bv1, nullptr, 1, nullptr, hv16);

    // 7. out = alpha * (hv @ V2 + bv2)  (M=1024, N=768, K=768)
    gemm_f16_kernel<<<dim3(HDIM / 128, SDIM / 64), 256, 0, stream>>>(
        hv16, V2h, SDIM, HDIM, HDIM, bv2, alpha, 0, (float*)d_out, nullptr);
}
